// HeadConvAttention_2302102470964
// MI455X (gfx1250) — compile-verified, matrix operand order checked by reading, not run on hardware
//
#include <hip/hip_runtime.h>

typedef __attribute__((ext_vector_type(16))) _Float16 v16h;
typedef __attribute__((ext_vector_type(8)))  float    v8f;
typedef __attribute__((ext_vector_type(2)))  float    v2f;

#define B_  4
#define H_  16
#define S_  1024
#define D_  64
#define QT  16
#define KT  32
#define L2E 1.44269504088896340736f

#define NKT16 (S_ / 16)   // 64 16-row k tiles
#define NKT32 (S_ / 32)   // 32 32-row k tiles

// Packed fragment workspace element counts (halfs).
#define WSK_ELEMS ((size_t)B_ * H_ * NKT16 * 2 * 32 * 16)   // 8 MiB
#define WSV_ELEMS ((size_t)B_ * H_ * NKT32 * 4 * 32 * 16)   // 8 MiB

#if __has_builtin(__builtin_amdgcn_wmma_f32_16x16x4_f32)
#define MIX_WMMA 1
#else
#define MIX_WMMA 0
#endif

// ---------------------------------------------------------------------------
// f16 16x32 A/B fragment from f32 row-major source.
// lane L: m(or n)=L%16, k-elements { kb..kb+7, kb+16..kb+23 }, kb=8*(L/16)
// ---------------------------------------------------------------------------
__device__ __forceinline__ v16h load_frag_f32(const float* __restrict__ src,
                                              int ld, int lane, float scale) {
  const int m  = lane & 15;
  const int kb = (lane >> 4) << 3;
  const float* p = src + (size_t)m * ld + kb;
  v16h f;
#pragma unroll
  for (int i = 0; i < 8; ++i) f[i]     = (_Float16)(p[i]      * scale);
#pragma unroll
  for (int i = 0; i < 8; ++i) f[i + 8] = (_Float16)(p[i + 16] * scale);
  return f;
}

__device__ __forceinline__ v16h load_frag_vcol(const float* __restrict__ vt,
                                               int lane, int d0) {
  const int n  = lane & 15;
  const int kb = (lane >> 4) << 3;
  v16h f;
#pragma unroll
  for (int i = 0; i < 8; ++i) f[i]     = (_Float16)vt[(size_t)(kb + i)      * D_ + d0 + n];
#pragma unroll
  for (int i = 0; i < 8; ++i) f[i + 8] = (_Float16)vt[(size_t)(kb + 16 + i) * D_ + d0 + n];
  return f;
}

__device__ __forceinline__ float redmax16(float v) {
#pragma unroll
  for (int m = 1; m <= 8; m <<= 1) v = fmaxf(v, __shfl_xor(v, m, 32));
  return v;
}
__device__ __forceinline__ float redsum16(float v) {
#pragma unroll
  for (int m = 1; m <= 8; m <<= 1) v += __shfl_xor(v, m, 32);
  return v;
}

// ---------------------------------------------------------------------------
// One-shot pre-pack kernels: K/V -> f16 fragments in exact WMMA lane order.
// ---------------------------------------------------------------------------
__global__ __launch_bounds__(256)
void prep_k_kernel(const float* __restrict__ xk, _Float16* __restrict__ wsK) {
  const int tid  = blockIdx.x * blockDim.x + threadIdx.x;
  const int lane = tid & 31;
  const int frag = tid >> 5;                 // bh*128 + kt*2 + c
  const int c    = frag & 1;
  const int kt   = (frag >> 1) & (NKT16 - 1);
  const int bh   = frag >> 7;
  const int n    = lane & 15;
  const int kb   = (lane >> 4) << 3;
  const float* p = xk + (((size_t)bh * S_) + kt * 16 + n) * D_ + c * 32 + kb;
  v16h f;
#pragma unroll
  for (int i = 0; i < 8; ++i) f[i]     = (_Float16)p[i];
#pragma unroll
  for (int i = 0; i < 8; ++i) f[i + 8] = (_Float16)p[i + 16];
  *(v16h*)(wsK + ((size_t)frag * 32 + lane) * 16) = f;
}

__global__ __launch_bounds__(256)
void prep_v_kernel(const float* __restrict__ xv, _Float16* __restrict__ wsV) {
  const int tid  = blockIdx.x * blockDim.x + threadIdx.x;
  const int lane = tid & 31;
  const int frag = tid >> 5;                 // bh*128 + kt32*4 + c
  const int c    = frag & 3;
  const int kt   = (frag >> 2) & (NKT32 - 1);
  const int bh   = frag >> 7;
  const int n    = lane & 15;
  const int kb   = (lane >> 4) << 3;
  const float* vt = xv + ((size_t)bh * S_ + kt * 32) * D_ + c * 16 + n;
  v16h f;
#pragma unroll
  for (int i = 0; i < 8; ++i) f[i]     = (_Float16)vt[(size_t)(kb + i)      * D_];
#pragma unroll
  for (int i = 0; i < 8; ++i) f[i + 8] = (_Float16)vt[(size_t)(kb + 16 + i) * D_];
  *(v16h*)(wsV + ((size_t)frag * 32 + lane) * 16) = f;
}

// ---------------------------------------------------------------------------
// Main head-mixed flash attention.
// ---------------------------------------------------------------------------
template <bool PACKED>
__global__ __launch_bounds__(512, 1)
void headconv_attn_kernel(const float* __restrict__ xq, const float* __restrict__ xk,
                          const float* __restrict__ xv, const float* __restrict__ W,
                          const int* __restrict__ causal_p, float* __restrict__ out,
                          const v16h* __restrict__ wsK, const v16h* __restrict__ wsV) {
  __shared__ float    lds_sc[H_][KT][QT];      // raw scores  [h][col][row], 32 KB
  __shared__ float    lds_mx[H_][KT][QT];      // mixed scores[g][col][row], 32 KB
  __shared__ _Float16 lds_p [H_][QT][KT];      // P staging   [h][row][col], 16 KB

  const int tid  = threadIdx.x;
  const int wv   = tid >> 5;      // wave == head
  const int lane = tid & 31;
  const int hl   = lane >> 4;
  const int cn   = lane & 15;
  const int b    = blockIdx.y;
  const int q0   = blockIdx.x * QT;
  const int causal = *causal_p;
  const int bh   = b * H_ + wv;

  const float* Qh = xq + (((size_t)bh) * S_ + q0) * D_;
  const float* Kh = xk + ((size_t)bh * S_) * D_;
  const float* Vh = xv + ((size_t)bh * S_) * D_;
  const v16h*  wsKh = PACKED ? (wsK + (size_t)bh * NKT16 * 2 * 32) : nullptr;
  const v16h*  wsVh = PACKED ? (wsV + (size_t)bh * NKT32 * 4 * 32) : nullptr;

#if MIX_WMMA
  // A-fragments of W for the f32 mixing WMMA (16x16x4, 4 K-chunks):
  // lane L: m = L%16, chunk ck holds K = 4*ck + 2*(L/16) + {0,1}
  v2f af[4];
#pragma unroll
  for (int ck = 0; ck < 4; ++ck) {
    af[ck][0] = W[cn * H_ + 4 * ck + 2 * hl + 0];
    af[ck][1] = W[cn * H_ + 4 * ck + 2 * hl + 1];
  }
#else
  float wrow[H_];
#pragma unroll
  for (int hh = 0; hh < H_; ++hh) wrow[hh] = W[wv * H_ + hh];
#endif

  // Q fragments, 1/sqrt(D)=0.125 folded in.
  v16h qf[2];
  qf[0] = load_frag_f32(Qh + 0,  D_, lane, 0.125f);
  qf[1] = load_frag_f32(Qh + 32, D_, lane, 0.125f);

  const v8f vzero = {};
  v8f o[4];
#pragma unroll
  for (int c = 0; c < 4; ++c) o[c] = vzero;

  float m_i[8], l_i[8];
#pragma unroll
  for (int j = 0; j < 8; ++j) { m_i[j] = -__builtin_inff(); l_i[j] = 0.0f; }

  const int kend = q0 + QT;
  for (int k0 = 0; k0 < kend; k0 += KT) {
    if (PACKED && (k0 + KT < kend)) {
      __builtin_prefetch(&wsKh[((size_t)((k0 + KT) >> 4) * 2) * 32 + lane], 0, 3);
      __builtin_prefetch(&wsVh[((size_t)((k0 + KT) >> 5) * 4) * 32 + lane], 0, 3);
    }

    // ---- K fragments (batched), then raw scores S = Q @ K^T (16x32) ----
    v16h kf[4];
#pragma unroll
    for (int t = 0; t < 2; ++t)
#pragma unroll
      for (int c = 0; c < 2; ++c) {
        if (PACKED)
          kf[t * 2 + c] = wsKh[((size_t)((k0 >> 4) + t) * 2 + c) * 32 + lane];
        else
          kf[t * 2 + c] = load_frag_f32(Kh + (size_t)(k0 + t * 16) * D_ + c * 32,
                                        D_, lane, 1.0f);
      }
    v8f s[2];
#pragma unroll
    for (int t = 0; t < 2; ++t) {
      v8f acc = vzero;
      acc = __builtin_amdgcn_wmma_f32_16x16x32_f16(false, qf[0], false, kf[t * 2 + 0],
                                                   (short)0, acc, false, false);
      acc = __builtin_amdgcn_wmma_f32_16x16x32_f16(false, qf[1], false, kf[t * 2 + 1],
                                                   (short)0, acc, false, false);
      s[t] = acc;
    }

    // ---- publish raw scores ([h][col][row]) ----
#pragma unroll
    for (int t = 0; t < 2; ++t)
#pragma unroll
      for (int j = 0; j < 8; ++j)
        lds_sc[wv][t * 16 + cn][hl * 8 + j] = s[t][j];

    // ---- V fragments issued early (consumed at end of iteration) ----
    v16h vf[4];
#pragma unroll
    for (int c = 0; c < 4; ++c) {
      if (PACKED)
        vf[c] = wsVh[((size_t)(k0 >> 5) * 4 + c) * 32 + lane];
      else
        vf[c] = load_frag_vcol(Vh + (size_t)k0 * D_, lane, c * 16);
    }

    __syncthreads();   // (A) raw scores visible

    float mx0[8], mx1[8];
#if MIX_WMMA
    // ---- head mixing on the matrix pipe: mixed = W @ raw (K = 16 heads) ----
    // wave wv owns local columns {2*wv, 2*wv+1}; computes all 16 output heads.
    const float* scb = &lds_sc[0][0][0];
#pragma unroll
    for (int t = 0; t < 2; ++t) {
      const int colp = 2 * wv + t;
      const int base = colp * QT + cn;
      v8f acc = vzero;
#pragma unroll
      for (int ck = 0; ck < 4; ++ck) {
        v2f bf;   // B 4x16: lane L holds n=L%16, K = 4*ck + 2*(L/16) + {0,1}
        bf[0] = scb[(size_t)(4 * ck + 2 * hl + 0) * (KT * QT) + base];
        bf[1] = scb[(size_t)(4 * ck + 2 * hl + 1) * (KT * QT) + base];
        acc = __builtin_amdgcn_wmma_f32_16x16x4_f32(false, af[ck], false, bf,
                                                    (short)0, acc, false, false);
      }
      // D: lane L holds g = j + 8*(L/16), r = L%16 -> scatter to [g][col][r]
#pragma unroll
      for (int j = 0; j < 8; ++j)
        lds_mx[j + 8 * hl][colp][cn] = acc[j];
    }
    __syncthreads();   // (B) mixed scores visible
    {
      const float* mp0 = &lds_mx[wv][cn][hl * 8];
      const float* mp1 = &lds_mx[wv][16 + cn][hl * 8];
#pragma unroll
      for (int j = 0; j < 8; ++j) { mx0[j] = mp0[j]; mx1[j] = mp1[j]; }
    }
#else
    // ---- VALU fallback mixing ----
#pragma unroll
    for (int j = 0; j < 8; ++j) { mx0[j] = 0.0f; mx1[j] = 0.0f; }
#pragma unroll
    for (int hh = 0; hh < H_; ++hh) {
      const float w = wrow[hh];
      const float* sp0 = &lds_sc[hh][0 * 16 + cn][hl * 8];
      const float* sp1 = &lds_sc[hh][1 * 16 + cn][hl * 8];
#pragma unroll
      for (int j = 0; j < 8; ++j) mx0[j] = fmaf(w, sp0[j], mx0[j]);
#pragma unroll
      for (int j = 0; j < 8; ++j) mx1[j] = fmaf(w, sp1[j], mx1[j]);
    }
    __syncthreads();   // (B)
#endif

    // ---- causal mask + online softmax ----
#pragma unroll
    for (int j = 0; j < 8; ++j) {
      const int row = q0 + hl * 8 + j;
      float s0 = mx0[j], s1 = mx1[j];
      if (causal) {
        if (k0 + cn      > row) s0 = -__builtin_inff();
        if (k0 + 16 + cn > row) s1 = -__builtin_inff();
      }
      const float rmax  = redmax16(fmaxf(s0, s1));
      const float mnew  = fmaxf(m_i[j], rmax);
      const float alpha = exp2f((m_i[j] - mnew) * L2E);
      const float p0    = exp2f((s0 - mnew) * L2E);
      const float p1    = exp2f((s1 - mnew) * L2E);
      const float rsum  = redsum16(p0 + p1);
      l_i[j] = l_i[j] * alpha + rsum;
      m_i[j] = mnew;
#pragma unroll
      for (int c = 0; c < 4; ++c) o[c][j] *= alpha;
      lds_p[wv][hl * 8 + j][cn]      = (_Float16)p0;
      lds_p[wv][hl * 8 + j][16 + cn] = (_Float16)p1;
    }

    // Wave-private region; DS ops in-order per wave -> no barrier needed.
    v16h pf;
    {
      const _Float16* pr = &lds_p[wv][cn][0];
      const int kb = hl * 8;
#pragma unroll
      for (int i = 0; i < 8; ++i) pf[i]     = pr[kb + i];
#pragma unroll
      for (int i = 0; i < 8; ++i) pf[i + 8] = pr[kb + 16 + i];
    }

    // ---- O += P @ V ----
#pragma unroll
    for (int c = 0; c < 4; ++c)
      o[c] = __builtin_amdgcn_wmma_f32_16x16x32_f16(false, pf, false, vf[c],
                                                    (short)0, o[c], false, false);
  }

  // ---- normalize + store out[b, q, h, d] ----
#pragma unroll
  for (int j = 0; j < 8; ++j) {
    const int row = q0 + hl * 8 + j;
    const float inv_l = 1.0f / l_i[j];
#pragma unroll
    for (int c = 0; c < 4; ++c) {
      const int d = c * 16 + cn;
      out[(((size_t)b * S_ + row) * H_ + wv) * D_ + d] = o[c][j] * inv_l;
    }
  }
}

extern "C" void kernel_launch(void* const* d_in, const int* in_sizes, int n_in,
                              void* d_out, int out_size, void* d_ws, size_t ws_size,
                              hipStream_t stream) {
  (void)in_sizes; (void)n_in; (void)out_size;
  const float* xq = (const float*)d_in[0];
  const float* xk = (const float*)d_in[1];
  const float* xv = (const float*)d_in[2];
  const float* W  = (const float*)d_in[3];
  const int* causal = (const int*)d_in[4];
  float* out = (float*)d_out;

  const dim3 grid(S_ / QT, B_);   // (64, 4)
  const dim3 block(H_ * 32);      // 16 waves, one head per wave

  const size_t need = (WSK_ELEMS + WSV_ELEMS) * sizeof(_Float16);  // 16 MiB
  if (ws_size >= need) {
    _Float16* wsK = (_Float16*)d_ws;
    _Float16* wsV = wsK + WSK_ELEMS;
    const int prep_threads = B_ * H_ * 128 * 32;   // 262144
    prep_k_kernel<<<prep_threads / 256, 256, 0, stream>>>(xk, wsK);
    prep_v_kernel<<<prep_threads / 256, 256, 0, stream>>>(xv, wsV);
    headconv_attn_kernel<true><<<grid, block, 0, stream>>>(
        xq, xk, xv, W, causal, out, (const v16h*)wsK, (const v16h*)wsV);
  } else {
    headconv_attn_kernel<false><<<grid, block, 0, stream>>>(
        xq, xk, xv, W, causal, out, nullptr, nullptr);
  }
}
